// RestrictedNN_62672162783264
// MI455X (gfx1250) — matrix operand order repaired and checked by reading, not compiled, not run on hardware
//
#include <hip/hip_runtime.h>
#include <hip/hip_bf16.h>

// ---------------------------------------------------------------------------
// Fused DCell-style ontology MLP for MI455X (gfx1250, wave32, WMMA).
//   kernel 1 (prep): pre-pack fp32 weights into per-lane f16 WMMA B-fragments
//   kernel 2 (fused): one pass over x; all intermediates live in LDS.
// ---------------------------------------------------------------------------

typedef __attribute__((ext_vector_type(16))) _Float16 v16h;
typedef __attribute__((ext_vector_type(8)))  float    v8f;

#define B_TOT   4096
#define L0_N    512
#define G_N     8
#define H_N     16
#define L1_N    64
#define L2_N    8

// workspace fragment regions (in halves). Each fragment = 32 lanes * 16 halves.
#define N0 (512 * 512)       // level-0: 512 leaves, 1 padded 32x16 frag each
#define N1 (64 * 4 * 512)    // level-1: 64 modules x 4 K-windows
#define N2 (8 * 4 * 512)     // level-2: 8 modules x 4 K-windows
#define N3 (4 * 512)         // root:    4 K-windows
#define NTOT (N0 + N1 + N2 + N3)

// LDS row strides (halves), chosen bank-conflict free (36m,68m,516m mod 64 distinct)
#define ST_G  72     // gene slab  [16][64]
#define ST_A1 136    // a1 concat  [16][128]
#define ST_H1 1032   // h1 concat  [16][1024]
#define ST_H2 136    // h2 concat  [16][128]

__device__ __forceinline__ unsigned short h16(float f) {
  _Float16 h = (_Float16)f;
  return __builtin_bit_cast(unsigned short, h);
}
__device__ __forceinline__ unsigned pack2(float a, float b) {
  return (unsigned)h16(a) | ((unsigned)h16(b) << 16);
}
__device__ __forceinline__ float sigmoidf(float v) {
  return 1.0f / (1.0f + __expf(-v));
}

// Gather a 16x32 f16 A-fragment from an LDS tile (row-major, 'stride' halves/row).
// A k order used here (matches prep's B packing): lane l holds row m=l&15; half t -> K = 16*(l>>4) + t.
__device__ __forceinline__ v16h gatherA(const unsigned short* base, int stride,
                                        int col0, int lane) {
  union { v16h v; unsigned u[8]; } A;
  const unsigned* p = (const unsigned*)base;
  const int m  = lane & 15;
  const int kb = (lane >> 4) * 16;
  const int rowoff = (m * stride + col0) >> 1;   // stride,col0 even
#pragma unroll
  for (int q = 0; q < 8; ++q) {                  // pair q covers K, K+1
    const int K = kb + 2 * q;
    A.u[q] = p[rowoff + (K >> 1)];
  }
  return A.v;
}

// sigmoid(D) -> f16 into LDS tile. C/D layout: lane l element i = D[i+8*(l>>4)][l&15]
__device__ __forceinline__ void storeSig(unsigned short* base, int stride,
                                         int col0, int lane, v8f c) {
  const int n  = lane & 15;
  const int mh = (lane >> 4) * 8;
#pragma unroll
  for (int i = 0; i < 8; ++i)
    base[(mh + i) * stride + col0 + n] = h16(sigmoidf(c[i]));
}

// ---------------------------------------------------------------------------
// Prep: pack weights into per-lane WMMA B-fragment order (f16).
// B k order used here (matches gatherA): lane l (n=l&15), half t -> K = 16*(l>>4) + t  (within 32-window)
// ---------------------------------------------------------------------------
__global__ __launch_bounds__(256) void prep_kernel(
    const float* __restrict__ W0, const float* __restrict__ W1,
    const float* __restrict__ W2, const float* __restrict__ W3,
    unsigned short* __restrict__ ws) {
  const int idx = blockIdx.x * 256 + threadIdx.x;
  if (idx >= NTOT) return;
  float val = 0.0f;
  if (idx < N0) {                        // level-0 padded frags
    const int L = idx >> 9, r = idx & 511;
    const int lane = r >> 4, t = r & 15;
    const int n = lane & 15;
    const int K = ((lane >> 4) << 4) + t;        // 0..31
    const int l3 = L & 7;                        // leaf within parent
    if ((K >> 3) == (l3 & 3))                    // 8 valid rows, rest zero
      val = W0[L * (G_N * H_N) + (K & 7) * H_N + n];
  } else if (idx < N0 + N1) {            // level-1
    const int r0 = idx - N0;
    const int mw = r0 >> 9, r = r0 & 511;
    const int j = mw >> 2, kw = mw & 3;
    const int lane = r >> 4, t = r & 15, n = lane & 15;
    const int K = kw * 32 + ((lane >> 4) << 4) + t;  // 0..127
    val = W1[j * 2048 + K * 16 + n];
  } else if (idx < N0 + N1 + N2) {       // level-2
    const int r0 = idx - (N0 + N1);
    const int mw = r0 >> 9, r = r0 & 511;
    const int j = mw >> 2, kw = mw & 3;
    const int lane = r >> 4, t = r & 15, n = lane & 15;
    const int K = kw * 32 + ((lane >> 4) << 4) + t;
    val = W2[j * 2048 + K * 16 + n];
  } else {                               // root
    const int r0 = idx - (N0 + N1 + N2);
    const int kw = r0 >> 9, r = r0 & 511;
    const int lane = r >> 4, t = r & 15, n = lane & 15;
    const int K = kw * 32 + ((lane >> 4) << 4) + t;
    val = W3[K * 16 + n];
  }
  ws[idx] = h16(val);
}

// ---------------------------------------------------------------------------
// Fused forward: one workgroup = 16 batch rows, 8 waves. Wave w owns L1
// modules 8w..8w+7 (x columns 512w..512w+511) and L2 module w.
// ---------------------------------------------------------------------------
__global__ __launch_bounds__(256) void fused_kernel(
    const float* __restrict__ x,  const float* __restrict__ Wg,
    const float* __restrict__ bg, const float* __restrict__ Wf,
    const unsigned short* __restrict__ ws, float* __restrict__ out) {
  __shared__ unsigned short aGene[8][16 * ST_G];   // per-wave gene slab (f16)
  __shared__ unsigned short a1s[8][16 * ST_A1];    // per-wave leaf concat (f16)
  __shared__ unsigned short h1buf[16 * ST_H1];     // workgroup h1 concat
  __shared__ unsigned short h2buf[16 * ST_H2];     // workgroup h2 concat
  __shared__ float hrbuf[16 * 16];                 // root partials

  const int lane = threadIdx.x & 31;
  const int wv   = threadIdx.x >> 5;
  const int r0   = blockIdx.x * 16;
  const int n    = lane & 15;
  const int mh   = (lane >> 4) * 8;

  unsigned short* ag = aGene[wv];
  unsigned short* a1 = a1s[wv];
  const v16h* wsB0 = (const v16h*)ws;                       // leaf*32 + lane
  const v16h* wsB1 = (const v16h*)(ws + N0);                // (j*4+kw)*32 + lane
  const v16h* wsB2 = (const v16h*)(ws + N0 + N1);           // (m2*4+kw)*32 + lane
  const v16h* wsB3 = (const v16h*)(ws + N0 + N1 + N2);      // kw*32 + lane

  for (int jj = 0; jj < 8; ++jj) {
    const int j = wv * 8 + jj;                    // L1 module id

    // ---- load x slab [16 rows x 64 genes], apply Wg/bg, stash f16 in LDS ----
#pragma unroll
    for (int t = 0; t < 4; ++t) {
      const int p = t * 32 + lane;                // (b,leaf) pair, 0..127
      const int b = p >> 3, leaf = p & 7;
      const float4* xp  = (const float4*)(x + (size_t)(r0 + b) * 4096 + j * 64 + leaf * 8);
      const float4* wgp = (const float4*)(Wg + j * 64 + leaf * 8);
      const float4* bgp = (const float4*)(bg + j * 64 + leaf * 8);
      const float4 x0 = xp[0],  x1 = xp[1];
      const float4 g0 = wgp[0], g1 = wgp[1];
      const float4 c0 = bgp[0], c1 = bgp[1];
      unsigned* dst = (unsigned*)ag;
      const int base = b * (ST_G / 2) + leaf * 4;
      dst[base + 0] = pack2(fmaf(x0.x, g0.x, c0.x), fmaf(x0.y, g0.y, c0.y));
      dst[base + 1] = pack2(fmaf(x0.z, g0.z, c0.z), fmaf(x0.w, g0.w, c0.w));
      dst[base + 2] = pack2(fmaf(x1.x, g1.x, c1.x), fmaf(x1.y, g1.y, c1.y));
      dst[base + 3] = pack2(fmaf(x1.z, g1.z, c1.z), fmaf(x1.w, g1.w, c1.w));
    }
    asm volatile("s_wait_dscnt 0" ::: "memory");

    // ---- level 0: 8 leaves, one K=32 WMMA each (8 valid K rows, padded) ----
#pragma unroll
    for (int l = 0; l < 8; ++l) {
      const v16h A = gatherA(ag, ST_G, 32 * (l >> 2), lane);
      const v16h Bm = wsB0[(j * 8 + l) * 32 + lane];
      v8f c = {};
      c = __builtin_amdgcn_wmma_f32_16x16x32_f16(false, A, false, Bm,
                                                 (short)0, c, false, false);
      storeSig(a1, ST_A1, l * 16, lane, c);       // a1 col = leaf*16 + h
    }
    asm volatile("s_wait_dscnt 0" ::: "memory");

    // ---- level 1: [16,128] @ [128,16], 4 chained WMMAs ----
    v8f c1 = {};
#pragma unroll
    for (int kw = 0; kw < 4; ++kw) {
      const v16h A = gatherA(a1, ST_A1, kw * 32, lane);
      const v16h Bm = wsB1[(j * 4 + kw) * 32 + lane];
      c1 = __builtin_amdgcn_wmma_f32_16x16x32_f16(false, A, false, Bm,
                                                  (short)0, c1, false, false);
    }
    storeSig(h1buf, ST_H1, j * 16, lane, c1);
  }
  __syncthreads();

  // ---- level 2: wave w computes module w from its own h1 columns ----
  {
    v8f c2 = {};
#pragma unroll
    for (int kw = 0; kw < 4; ++kw) {
      const v16h A = gatherA(h1buf, ST_H1, wv * 128 + kw * 32, lane);
      const v16h Bm = wsB2[(wv * 4 + kw) * 32 + lane];
      c2 = __builtin_amdgcn_wmma_f32_16x16x32_f16(false, A, false, Bm,
                                                  (short)0, c2, false, false);
    }
    storeSig(h2buf, ST_H2, wv * 16, lane, c2);
  }
  __syncthreads();

  // ---- root + final dense (wave 0) ----
  if (wv == 0) {
    v8f c3 = {};
#pragma unroll
    for (int kw = 0; kw < 4; ++kw) {
      const v16h A = gatherA(h2buf, ST_H2, kw * 32, lane);
      const v16h Bm = wsB3[kw * 32 + lane];
      c3 = __builtin_amdgcn_wmma_f32_16x16x32_f16(false, A, false, Bm,
                                                  (short)0, c3, false, false);
    }
    const float wf = Wf[n];
#pragma unroll
    for (int i = 0; i < 8; ++i)
      hrbuf[(mh + i) * 16 + n] = sigmoidf(c3[i]) * wf;
    asm volatile("s_wait_dscnt 0" ::: "memory");
    if (lane < 16) {
      float s = 0.0f;
#pragma unroll
      for (int nn = 0; nn < 16; ++nn) s += hrbuf[lane * 16 + nn];
      out[r0 + lane] = s;
    }
  }
}

// ---------------------------------------------------------------------------
extern "C" void kernel_launch(void* const* d_in, const int* in_sizes, int n_in,
                              void* d_out, int out_size, void* d_ws, size_t ws_size,
                              hipStream_t stream) {
  if (n_in < 8) return;
  const float* x  = (const float*)d_in[0];
  const float* Wg = (const float*)d_in[1];
  const float* bg = (const float*)d_in[2];
  const float* W0 = (const float*)d_in[3];
  const float* W1 = (const float*)d_in[4];
  const float* W2 = (const float*)d_in[5];
  const float* W3 = (const float*)d_in[6];
  const float* Wf = (const float*)d_in[7];
  unsigned short* ws = (unsigned short*)d_ws;
  float* out = (float*)d_out;

  prep_kernel<<<(NTOT + 255) / 256, 256, 0, stream>>>(W0, W1, W2, W3, ws);
  fused_kernel<<<B_TOT / 16, 256, 0, stream>>>(x, Wg, bg, Wf, ws, out);
}
